// ReinforcementLearningBase_20933670600845
// MI455X (gfx1250) — compile-verified, matrix operand order checked by reading, not run on hardware
//
#include <hip/hip_runtime.h>
#include <hip/hip_bf16.h>
#include <math.h>

// ---------------------------------------------------------------------------
// Problem constants (from the reference)
// ---------------------------------------------------------------------------
#define S_LEN   50     // outer sequence length
#define BATCH   256
#define IN_SZ   200
#define EMB     80
#define EPAD    96     // EMB padded to multiple of 32 for WMMA K
#define MEMN    256    // memory size M
#define TSTEPS  50     // inner simulation steps
#define TPAD    64     // TSTEPS padded to multiple of 32
#define RWIN    30
#define A_DEC   0.9f
#define ETA     0.01f
#define LAM     0.9512294245007140f   // exp(-1/20)

typedef __attribute__((ext_vector_type(16))) __bf16 v16bf;
typedef __attribute__((ext_vector_type(8)))  __bf16 v8bf;
typedef __attribute__((ext_vector_type(8)))  float  v8f;

// ---------------------------------------------------------------------------
// LIF dynamics (soft spike surrogate, threshold 1.0)
// ---------------------------------------------------------------------------
__device__ __forceinline__ void lif_step(float& ci, float& cv, float& cz, float inp) {
  ci = A_DEC * ci + inp;
  cv = A_DEC * cv * (1.0f - cz) + ci;
  cz = 1.0f / (1.0f + __expf(-10.0f * (cv - 1.0f)));
}

// ---------------------------------------------------------------------------
// WMMA fragment load, K-contiguous source.
// 16-bit A-matrix 16x32 striping (CDNA5 ISA):
//   lanes 0-15 : M = lane,     half-slot j -> K = (j<8 ? j : j+8)
//   lanes 16-31: M = lane-16,  half-slot j -> K = (j<8 ? j : j+8) + 8
// With the lane-group base folded into the pointer, slots 0..7 are p[0..7]
// and slots 8..15 are p[16..23]: two aligned 16-byte loads.
// B operand is supplied as B^T in the same striping (lane = N column).
// ---------------------------------------------------------------------------
__device__ __forceinline__ v16bf load_frag(const __bf16* __restrict__ p) {
  v8bf lo = *(const v8bf*)(p);
  v8bf hi = *(const v8bf*)(p + 16);
  v16bf f;
#pragma unroll
  for (int j = 0; j < 8; ++j) { f[j] = lo[j]; f[j + 8] = hi[j]; }
  return f;
}

// ---------------------------------------------------------------------------
// Batched bf16->f32 WMMA GEMM:  C[b] = A[b] (Mrows x K) * B[b] (K x Ncols)
// B given transposed (Bt: Ncols x K).  All dims multiples of 32, K % 32 == 0,
// all operands K-contiguous (unit K stride), rows 16B-aligned.
// One wave per 32x32 output tile: 2x2 WMMA sub-tiles, 4 accumulators.
// Each K-step: 8x global_load_b128 -> 4x v_wmma (A/B fragments reused twice).
// mode 0: C[idx]  = acc
// mode 1: C[idx] += scale*acc ; shadow[idx] = bf16(C[idx])   (memory update)
// ---------------------------------------------------------------------------
__global__ void wmma_gemm_bf16(const __bf16* __restrict__ A, long a_rs, long a_bs,
                               const __bf16* __restrict__ Bt, long b_rs, long b_bs,
                               float* __restrict__ C, long c_ld, long c_bs,
                               int K, int mode, float scale,
                               __bf16* __restrict__ shadow) {
  const int lane    = threadIdx.x & 31;
  const int lane_hi = lane >> 4;
  const int tileMp  = blockIdx.x;   // 32-row output stripe
  const int tileNp  = blockIdx.y;   // 32-col output stripe
  const int b       = blockIdx.z;

  const int rowA = tileMp * 32 + (lane & 15);
  const int rowB = tileNp * 32 + (lane & 15);
  const __bf16* pa0 = A  + (long)b * a_bs + (long)rowA * a_rs + (lane_hi ? 8 : 0);
  const __bf16* pa1 = pa0 + 16 * a_rs;
  const __bf16* pb0 = Bt + (long)b * b_bs + (long)rowB * b_rs + (lane_hi ? 8 : 0);
  const __bf16* pb1 = pb0 + 16 * b_rs;

  v8f acc00 = {0.f,0.f,0.f,0.f,0.f,0.f,0.f,0.f};
  v8f acc01 = acc00, acc10 = acc00, acc11 = acc00;

  for (int k0 = 0; k0 < K; k0 += 32) {
    v16bf fa0 = load_frag(pa0 + k0);
    v16bf fa1 = load_frag(pa1 + k0);
    v16bf fb0 = load_frag(pb0 + k0);
    v16bf fb1 = load_frag(pb1 + k0);
    acc00 = __builtin_amdgcn_wmma_f32_16x16x32_bf16(false, fa0, false, fb0,
                                                    (short)0, acc00, false, false);
    acc01 = __builtin_amdgcn_wmma_f32_16x16x32_bf16(false, fa0, false, fb1,
                                                    (short)0, acc01, false, false);
    acc10 = __builtin_amdgcn_wmma_f32_16x16x32_bf16(false, fa1, false, fb0,
                                                    (short)0, acc10, false, false);
    acc11 = __builtin_amdgcn_wmma_f32_16x16x32_bf16(false, fa1, false, fb1,
                                                    (short)0, acc11, false, false);
  }

  // C/D layout: lanes 0-15: N=lane, M=r ; lanes 16-31: N=lane-16, M=8+r
  const int n0    = tileNp * 32 + (lane & 15);
  const int mb0   = tileMp * 32 + (lane_hi ? 8 : 0);
  const long base = (long)b * c_bs + (long)mb0 * c_ld + n0;
  if (mode == 0) {
#pragma unroll
    for (int r = 0; r < 8; ++r) {
      long row = base + (long)r * c_ld;
      C[row]                  = acc00[r];
      C[row + 16]             = acc01[r];
      C[row + 16 * c_ld]      = acc10[r];
      C[row + 16 * c_ld + 16] = acc11[r];
    }
  } else {
#pragma unroll
    for (int r = 0; r < 8; ++r) {
      long row = base + (long)r * c_ld;
      long i00 = row, i01 = row + 16, i10 = row + 16 * c_ld, i11 = i10 + 16;
      float v00 = C[i00] + scale * acc00[r];
      float v01 = C[i01] + scale * acc01[r];
      float v10 = C[i10] + scale * acc10[r];
      float v11 = C[i11] + scale * acc11[r];
      C[i00] = v00; C[i01] = v01; C[i10] = v10; C[i11] = v11;
      shadow[i00] = (__bf16)v00; shadow[i01] = (__bf16)v01;
      shadow[i10] = (__bf16)v10; shadow[i11] = (__bf16)v11;
    }
  }
}

// ---------------------------------------------------------------------------
// Small helpers / elementwise kernels
// ---------------------------------------------------------------------------
__global__ void zeroK(float* __restrict__ p, long n) {
  long i = (long)blockIdx.x * blockDim.x + threadIdx.x;
  if (i < n) p[i] = 0.0f;
}

__global__ void init_mem(const float* __restrict__ mem0, float* __restrict__ memf,
                         __bf16* __restrict__ shadow, long n) {
  long i = (long)blockIdx.x * blockDim.x + threadIdx.x;
  if (i < n) {
    float v = mem0[i];
    memf[i]   = v;
    shadow[i] = (__bf16)v;
  }
}

// fp32 [MEMN, EMB] -> bf16 [MEMN, EPAD] (zero padded)
__global__ void conv_pad_w(const float* __restrict__ W, __bf16* __restrict__ Wb) {
  int i = blockIdx.x * blockDim.x + threadIdx.x;
  if (i >= MEMN * EPAD) return;
  int r = i / EPAD, e = i % EPAD;
  Wb[i] = (e < EMB) ? (__bf16)W[r * EMB + e] : (__bf16)0.0f;
}

// tmp[b,e] = sum_i facts[b,i] * W_emb[e,i]
__global__ void emb_proj(const float* __restrict__ x, const float* __restrict__ Wemb,
                         float* __restrict__ tmp) {
  int i = blockIdx.x * blockDim.x + threadIdx.x;
  if (i >= BATCH * EMB) return;
  int b = i / EMB, e = i % EMB;
  const float* xr = x + (long)b * IN_SZ;
  const float* wr = Wemb + (long)e * IN_SZ;
  float s = 0.f;
  for (int k = 0; k < IN_SZ; ++k) s += xr[k] * wr[k];
  tmp[i] = s;
}

// enc_in[b,e] = sum_j tmp[b,j] * W_enc[e,j]
__global__ void enc_proj(const float* __restrict__ tmp, const float* __restrict__ Wenc,
                         float* __restrict__ enc_in) {
  int i = blockIdx.x * blockDim.x + threadIdx.x;
  if (i >= BATCH * EMB) return;
  int b = i / EMB, e = i % EMB;
  const float* tr = tmp + (long)b * EMB;
  const float* wr = Wenc + (long)e * EMB;
  float s = 0.f;
  for (int k = 0; k < EMB; ++k) s += tr[k] * wr[k];
  enc_in[i] = s;
}

// Encoder LIF over B x EPAD: pad lanes (e >= EMB) write zeros so the
// downstream GEMM needs no K bounds.  State persists in ws across outer steps.
__global__ void enc_lif(const float* __restrict__ enc_in, float* __restrict__ st,
                        __bf16* __restrict__ Ze) {
  const int BE = BATCH * EMB;
  int i = blockIdx.x * blockDim.x + threadIdx.x;
  if (i >= BATCH * EPAD) return;
  int b = i / EPAD, e = i % EPAD;
  if (e >= EMB) {
    for (int t = 0; t < TSTEPS; ++t)
      Ze[((long)t * BATCH + b) * EPAD + e] = (__bf16)0.0f;
    return;
  }
  int be = b * EMB + e;
  float ci = st[be], cv = st[BE + be], cz = st[2 * BE + be];
  float inp = enc_in[be];
  for (int t = 0; t < TSTEPS; ++t) {
    lif_step(ci, cv, cz, inp);
    Ze[((long)t * BATCH + b) * EPAD + e] = (__bf16)cz;
  }
  st[be] = ci; st[BE + be] = cv; st[2 * BE + be] = cz;
}

// wk/wv/rk LIFs + eligibility traces.  Emits K-contiguous layouts:
//   Tkh, Zrk : [TPAD, B, M]  (K = memory index; pad rows zeroed)
//   ZvT, TkT : [B, M, TPAD]  (K = time; pad cols zeroed)
__global__ void mid_lif(const float* __restrict__ Iwk, const float* __restrict__ Iwv,
                        const float* __restrict__ Irk, float* __restrict__ st,
                        __bf16* __restrict__ Tkh, __bf16* __restrict__ Zrk,
                        __bf16* __restrict__ ZvT, __bf16* __restrict__ TkT) {
  const long BM = (long)BATCH * MEMN;
  long i = (long)blockIdx.x * blockDim.x + threadIdx.x;
  if (i >= BM) return;
  int b = (int)(i / MEMN), m = (int)(i % MEMN);
  float wki = st[0*BM+i], wkv = st[1*BM+i], wkz = st[2*BM+i];
  float wvi = st[3*BM+i], wvv = st[4*BM+i], wvz = st[5*BM+i];
  float rki = st[6*BM+i], rkv = st[7*BM+i], rkz = st[8*BM+i];
  float tk  = st[9*BM+i], tv  = st[10*BM+i];
  for (int t = 0; t < TSTEPS; ++t) {
    long ridx = ((long)t * BATCH + b) * MEMN + m;
    lif_step(wki, wkv, wkz, Iwk[ridx]);
    lif_step(wvi, wvv, wvz, Iwv[ridx]);
    tk = LAM * tk + wkz;
    tv = LAM * tv + wvz;
    lif_step(rki, rkv, rkz, Irk[ridx]);
    Tkh[(long)t * BM + i] = (__bf16)tk;
    Zrk[(long)t * BM + i] = (__bf16)rkz;
    ZvT[i * TPAD + t]     = (__bf16)wvz;
    TkT[i * TPAD + t]     = (__bf16)tk;
  }
  for (int t = TSTEPS; t < TPAD; ++t) {   // zero K/N pads
    Tkh[(long)t * BM + i] = (__bf16)0.0f;
    Zrk[(long)t * BM + i] = (__bf16)0.0f;
    ZvT[i * TPAD + t]     = (__bf16)0.0f;
    TkT[i * TPAD + t]     = (__bf16)0.0f;
  }
  st[0*BM+i]=wki; st[1*BM+i]=wkv; st[2*BM+i]=wkz;
  st[3*BM+i]=wvi; st[4*BM+i]=wvv; st[5*BM+i]=wvz;
  st[6*BM+i]=rki; st[7*BM+i]=rkv; st[8*BM+i]=rkz;
  st[9*BM+i]=tk;  st[10*BM+i]=tv;
}

// Mask (s<=t, both < TSTEPS), scale by ETA, transpose to GmT[b][t][s] (bf16)
// so the correction GEMM's B operand is K(s)-contiguous.
__global__ void g_mask(const float* __restrict__ Graw, __bf16* __restrict__ GmT) {
  int i = blockIdx.x * blockDim.x + threadIdx.x;
  if (i >= BATCH * TPAD * TPAD) return;
  int b = i / (TPAD * TPAD);
  int r = i % (TPAD * TPAD);
  int t = r / TPAD, s = r % TPAD;
  float g = 0.0f;
  if (s <= t && s < TSTEPS && t < TSTEPS)
    g = ETA * Graw[(long)b * TPAD * TPAD + (long)s * TPAD + t];
  GmT[(long)b * TPAD * TPAD + (long)t * TPAD + s] = (__bf16)g;
}

// Read-value LIF + delayed readout + last-RWIN sum -> output row for step s.
__global__ void rv_out(const float* __restrict__ R0, const float* __restrict__ Corr,
                       float* __restrict__ st, float* __restrict__ out) {
  const long BM = (long)BATCH * MEMN;
  long i = (long)blockIdx.x * blockDim.x + threadIdx.x;
  if (i >= BM) return;
  float ci = st[0*BM+i], cv = st[1*BM+i], cz = st[2*BM+i], buf = st[3*BM+i];
  const long base = i * TPAD;    // [B, M, TPAD] flat
  float acc = 0.f;
  for (int t = 0; t < TSTEPS; ++t) {
    float rin = R0[base + t] + Corr[base + t];
    lif_step(ci, cv, cz, rin);
    if (t >= TSTEPS - RWIN) acc += buf;   // readout delayed by one inner step
    buf = cz;
  }
  out[i] = acc;
  st[0*BM+i]=ci; st[1*BM+i]=cv; st[2*BM+i]=cz; st[3*BM+i]=buf;
}

// ---------------------------------------------------------------------------
// Host orchestration (graph-capture safe: stream launches only)
// ---------------------------------------------------------------------------
extern "C" void kernel_launch(void* const* d_in, const int* in_sizes, int n_in,
                              void* d_out, int out_size, void* d_ws, size_t ws_size,
                              hipStream_t stream) {
  const float* facts = (const float*)d_in[0];   // [S, B, I]
  const float* mem0  = (const float*)d_in[1];   // [B, M, M]
  const float* Wemb  = (const float*)d_in[2];   // [E, I]
  const float* Wenc  = (const float*)d_in[3];   // [E, E]
  const float* Wwk   = (const float*)d_in[4];   // [M, E]
  const float* Wwv   = (const float*)d_in[5];
  const float* Wrk   = (const float*)d_in[6];
  float* out = (float*)d_out;                   // [S, B, M]

  const long BM   = (long)BATCH * MEMN;         // 65536
  const long BE   = (long)BATCH * EMB;          // 20480
  const long TB   = (long)TSTEPS * BATCH;       // 12800
  const long MM   = (long)MEMN * MEMN;
  const long NMEM = (long)BATCH * MM;           // 16.7M
  const long TT   = (long)TPAD * TPAD;

  // --- bump allocator over d_ws -------------------------------------------
  size_t off = 0;
  auto alloc = [&](size_t bytes) -> void* {
    void* p = (char*)d_ws + off;
    off += (bytes + 255) & ~(size_t)255;
    return p;
  };
  float*  memf   = (float*) alloc(NMEM * 4);            // fp32 master memory
  __bf16* shadow = (__bf16*)alloc(NMEM * 2);            // bf16 shadow for WMMA reads
  __bf16* Ze     = (__bf16*)alloc(TB * EPAD * 2);       // [T*B, EPAD]
  float*  Iwk    = (float*) alloc(TB * MEMN * 4);
  float*  Iwv    = (float*) alloc(TB * MEMN * 4);
  float*  Irk    = (float*) alloc(TB * MEMN * 4);
  __bf16* Tkh    = (__bf16*)alloc((long)TPAD * BM * 2); // [TPAD, B, M]
  __bf16* Zrk    = (__bf16*)alloc((long)TPAD * BM * 2); // [TPAD, B, M]
  __bf16* ZvT    = (__bf16*)alloc(BM * TPAD * 2);       // [B, M, TPAD]
  __bf16* TkT    = (__bf16*)alloc(BM * TPAD * 2);       // [B, M, TPAD]
  float*  R0     = (float*) alloc(BM * TPAD * 4);       // [B, M, TPAD]
  float*  Corr   = (float*) alloc(BM * TPAD * 4);       // [B, M, TPAD]
  float*  Graw   = (float*) alloc((long)BATCH * TT * 4);
  __bf16* GmT    = (__bf16*)alloc((long)BATCH * TT * 2);
  __bf16* WwkB   = (__bf16*)alloc((long)MEMN * EPAD * 2);
  __bf16* WwvB   = (__bf16*)alloc((long)MEMN * EPAD * 2);
  __bf16* WrkB   = (__bf16*)alloc((long)MEMN * EPAD * 2);
  float*  tmpE   = (float*) alloc(BE * 4);
  float*  encIn  = (float*) alloc(BE * 4);
  float*  encSt  = (float*) alloc(3 * BE * 4);
  float*  midSt  = (float*) alloc(11 * BM * 4);
  float*  rvSt   = (float*) alloc(4 * BM * 4);
  (void)ws_size; (void)in_sizes; (void)n_in; (void)out_size;

  // --- per-call (deterministic) initialization ----------------------------
  init_mem<<<(unsigned)((NMEM + 255) / 256), 256, 0, stream>>>(mem0, memf, shadow, NMEM);
  zeroK<<<(unsigned)((3 * BE + 255) / 256), 256, 0, stream>>>(encSt, 3 * BE);
  zeroK<<<(unsigned)((11 * BM + 255) / 256), 256, 0, stream>>>(midSt, 11 * BM);
  zeroK<<<(unsigned)((4 * BM + 255) / 256), 256, 0, stream>>>(rvSt, 4 * BM);
  conv_pad_w<<<(MEMN * EPAD + 255) / 256, 256, 0, stream>>>(Wwk, WwkB);
  conv_pad_w<<<(MEMN * EPAD + 255) / 256, 256, 0, stream>>>(Wwv, WwvB);
  conv_pad_w<<<(MEMN * EPAD + 255) / 256, 256, 0, stream>>>(Wrk, WrkB);

  const unsigned beBlocks  = (unsigned)((BE + 255) / 256);
  const unsigned bepBlocks = (unsigned)((BATCH * EPAD + 255) / 256);
  const unsigned bmBlocks  = (unsigned)((BM + 255) / 256);

  for (int s = 0; s < S_LEN; ++s) {
    const float* x_t = facts + (long)s * BATCH * IN_SZ;

    // enc_in = (x @ Wemb^T) @ Wenc^T
    emb_proj<<<beBlocks, 256, 0, stream>>>(x_t, Wemb, tmpE);
    enc_proj<<<beBlocks, 256, 0, stream>>>(tmpE, Wenc, encIn);

    // encoder spike history Ze[T, B, EPAD] (pads zeroed)
    enc_lif<<<bepBlocks, 256, 0, stream>>>(encIn, encSt, Ze);

    // Iwk/Iwv/Irk = Ze @ W^T  — [12800,96]x[96,256] WMMA GEMM (32x32 waves)
    dim3 g2((unsigned)(TB / 32), MEMN / 32, 1);
    wmma_gemm_bf16<<<g2, 32, 0, stream>>>(Ze, EPAD, 0, WwkB, EPAD, 0,
                                          Iwk, MEMN, 0, EPAD, 0, 1.0f, nullptr);
    wmma_gemm_bf16<<<g2, 32, 0, stream>>>(Ze, EPAD, 0, WwvB, EPAD, 0,
                                          Iwv, MEMN, 0, EPAD, 0, 1.0f, nullptr);
    wmma_gemm_bf16<<<g2, 32, 0, stream>>>(Ze, EPAD, 0, WrkB, EPAD, 0,
                                          Irk, MEMN, 0, EPAD, 0, 1.0f, nullptr);

    // wk/wv/rk LIFs + traces -> Tkh, Zrk, ZvT, TkT
    mid_lif<<<bmBlocks, 256, 0, stream>>>(Iwk, Iwv, Irk, midSt, Tkh, Zrk, ZvT, TkT);

    // R0[b] = mem_enter[b] @ Zrk[b]^T : A=shadow[b](M,M), Bt[t] = Zrk[t,b,:]
    dim3 g4(MEMN / 32, TPAD / 32, BATCH);
    wmma_gemm_bf16<<<g4, 32, 0, stream>>>(shadow, MEMN, MM,
                                          Zrk, BM, MEMN,
                                          R0, TPAD, (long)MEMN * TPAD,
                                          MEMN, 0, 1.0f, nullptr);

    // Graw[b][s][t] = Tk_s . Zrk_t : A[s]=Tkh[s,b,:], Bt[t]=Zrk[t,b,:], K=M
    dim3 g5(TPAD / 32, TPAD / 32, BATCH);
    wmma_gemm_bf16<<<g5, 32, 0, stream>>>(Tkh, BM, MEMN,
                                          Zrk, BM, MEMN,
                                          Graw, TPAD, TT,
                                          MEMN, 0, 1.0f, nullptr);
    g_mask<<<(unsigned)((BATCH * TT + 255) / 256), 256, 0, stream>>>(Graw, GmT);

    // Corr[b][m][t] = sum_s ZvT[b,m,s] * GmT[b,t,s] : K = TPAD
    dim3 g6(MEMN / 32, TPAD / 32, BATCH);
    wmma_gemm_bf16<<<g6, 32, 0, stream>>>(ZvT, TPAD, (long)MEMN * TPAD,
                                          GmT, TPAD, TT,
                                          Corr, TPAD, (long)MEMN * TPAD,
                                          TPAD, 0, 1.0f, nullptr);

    // mem += ETA * Zv^T @ Tk : A[m]=ZvT[b,m,:], Bt[n]=TkT[b,n,:], K=TPAD
    dim3 g7(MEMN / 32, MEMN / 32, BATCH);
    wmma_gemm_bf16<<<g7, 32, 0, stream>>>(ZvT, TPAD, (long)MEMN * TPAD,
                                          TkT, TPAD, (long)MEMN * TPAD,
                                          memf, MEMN, MM,
                                          TPAD, 1, ETA, shadow);

    // rv LIF + delayed readout + last-30 sum -> out[s]
    rv_out<<<bmBlocks, 256, 0, stream>>>(R0, Corr, rvSt, out + (long)s * BM);
  }
}
